// TextGATConv_mod_37434934952836
// MI455X (gfx1250) — matrix operand order repaired and checked by reading, not compiled, not run on hardware
//
#include <hip/hip_runtime.h>
#include <hip/hip_bf16.h>

typedef __attribute__((ext_vector_type(16))) _Float16 v16h;
typedef __attribute__((ext_vector_type(8)))  float    v8f;

#define BB 8
#define LL 128
#define DD 512
#define EE 64
#define HH 8
#define CC 64

union V16H { v16h v; uint4 u[2]; _Float16 h[16]; };
union U8H  { uint4 u; _Float16 h[8]; };
union U4H  { uint2 u; _Float16 h[4]; };

// ---- WMMA fragment builders (per CDNA5 ISA 7.12.2 layouts, wave32) ----

// A 16x32 f16 (row-major source): lane<16 -> K{0..7,16..23}, lane>=16 -> K{8..15,24..31}.
__device__ __forceinline__ v16h a_frag_from(const _Float16* rowptr, int lane) {
    const int kb = (lane & 16) ? 8 : 0;
    V16H t;
    t.u[0] = *(const uint4*)(rowptr + kb);
    t.u[1] = *(const uint4*)(rowptr + 16 + kb);
    return t.v;
}

// B 32x16 f16 from TRANSPOSED weight (N x K row-major): lane<16 -> K{0..7,16..23},
// lane>=16 -> K{8..15,24..31}. p = Bt + col*ldk + k0.
__device__ __forceinline__ v16h b_frag_t(const _Float16* p, int lane) {
    const int kb = (lane & 16) ? 8 : 0;
    V16H t;
    t.u[0] = *(const uint4*)(p + kb);
    t.u[1] = *(const uint4*)(p + 16 + kb);
    return t.v;
}

__device__ __forceinline__ v8f wmma_f16(v16h a, v16h b, v8f c) {
    return __builtin_amdgcn_wmma_f32_16x16x32_f16(false, a, false, b, (short)0, c, false, false);
}

// ---- f32 -> f16 convert ----
__global__ void cvt_f32_f16(const float* __restrict__ in, _Float16* __restrict__ out, int n) {
    int i = blockIdx.x * blockDim.x + threadIdx.x;
    if (i < n) out[i] = (_Float16)in[i];
}

// ---- f32 (K x N) -> f16 transposed (N x K) ----
__global__ void cvt_transpose_f16(const float* __restrict__ in, _Float16* __restrict__ out,
                                  int K, int N) {
    int i = blockIdx.x * blockDim.x + threadIdx.x;
    if (i < K * N) {
        int k = i / N, n = i - k * N;
        out[(size_t)n * K + k] = (_Float16)in[i];
    }
}

// ---- f16 GEMM: C(MxN,f32) = A(MxK,row-major) @ Bt(NxK,row-major = B transposed) ----
// block: 256 thr (8 waves), tile 128x64, K-step 32. wave -> 32x32 (2x2 subtiles).
// A staged through LDS (straight-line copy, loads hoisted above stores);
// B fragments straight from global (L2-resident).
__global__ __launch_bounds__(256) void gemm_f16(const _Float16* __restrict__ A,
                                                const _Float16* __restrict__ Bt,
                                                float* __restrict__ C,
                                                int M, int N, int K) {
    __shared__ _Float16 As[128][48];   // 32 halves payload + pad, rows 16B aligned
    const int t = threadIdx.x, lane = t & 31, wave = t >> 5;
    const int m0 = blockIdx.y * 128, n0 = blockIdx.x * 64;
    const int wm = wave >> 1, wn = wave & 1;
    const int coln0 = n0 + wn * 32 + (lane & 15);

    v8f acc[2][2];
    v8f zero = {};
#pragma unroll
    for (int i = 0; i < 2; ++i)
#pragma unroll
        for (int j = 0; j < 2; ++j) acc[i][j] = zero;

    const int r0 = t >> 2, c0 = (t & 3) * 8;   // staging: 2 fixed uint4 per thread
    const int r1 = r0 + 64;

    for (int k0 = 0; k0 < K; k0 += 32) {
        const _Float16* s0 = &A[(size_t)(m0 + r0) * K + k0 + c0];
        const _Float16* s1 = &A[(size_t)(m0 + r1) * K + k0 + c0];
        uint4 d0 = *(const uint4*)s0;
        uint4 d1 = *(const uint4*)s1;
        if (k0 + 32 < K) {
            __builtin_prefetch(s0 + 32, 0, 3);
            __builtin_prefetch(s1 + 32, 0, 3);
        }
        *(uint4*)&As[r0][c0] = d0;
        *(uint4*)&As[r1][c0] = d1;
        __syncthreads();
        v16h a0 = a_frag_from(&As[wm * 32 + 0  + (lane & 15)][0], lane);
        v16h a1 = a_frag_from(&As[wm * 32 + 16 + (lane & 15)][0], lane);
        v16h b0 = b_frag_t(Bt + (size_t)(coln0 +  0) * K + k0, lane);
        v16h b1 = b_frag_t(Bt + (size_t)(coln0 + 16) * K + k0, lane);
        acc[0][0] = wmma_f16(a0, b0, acc[0][0]);
        acc[0][1] = wmma_f16(a0, b1, acc[0][1]);
        acc[1][0] = wmma_f16(a1, b0, acc[1][0]);
        acc[1][1] = wmma_f16(a1, b1, acc[1][1]);
        __syncthreads();
    }
    const int roff = (lane & 16) ? 8 : 0;
#pragma unroll
    for (int ms = 0; ms < 2; ++ms)
#pragma unroll
        for (int ns = 0; ns < 2; ++ns)
#pragma unroll
            for (int r = 0; r < 8; ++r) {
                int row = m0 + wm * 32 + ms * 16 + roff + r;
                int col = n0 + wn * 32 + ns * 16 + (lane & 15);
                C[(size_t)row * N + col] = acc[ms][ns][r];
            }
}

// ---- fused edge-GAT attention ----
// grid: (L/16, H, B); block 256 (8 waves). Per 16-key tile:
//  kve(256 pairs x 64c) = e_tile(256x64E) @ We + kv  -- kv folded into the WMMA
//  C-operand (accumulator init), result spilled to LDS as f16; then
//  score/leaky/mask/online-softmax/weighted-sum with half-wave shuffles.
__global__ __launch_bounds__(256) void attn_fused(const _Float16* __restrict__ e_h,
                                                  const _Float16* __restrict__ We_t, // (D x E)
                                                  const float* __restrict__ q_f,
                                                  const float* __restrict__ kv_f,
                                                  const int* __restrict__ adj,
                                                  _Float16* __restrict__ attn_h) {
    __shared__ _Float16 ee_s[256][64];   // kve tile, f16 (32 KB)
    __shared__ float q_s[16][64];
    __shared__ float kv_s[16][64];
    const int t = threadIdx.x, lane = t & 31, wave = t >> 5;
    const int qt = blockIdx.x, h = blockIdx.y, b = blockIdx.z;
    const int q0 = qt * 16;
    const int roff = (lane & 16) ? 8 : 0;

    // preload B fragments of We slice for head h from transposed weight: 4 Nsub x 2 Ksteps
    v16h bfr[4][2];
#pragma unroll
    for (int ks = 0; ks < 2; ++ks)
#pragma unroll
        for (int n = 0; n < 4; ++n) {
            int col = h * CC + n * 16 + (lane & 15);        // output channel d
            bfr[n][ks] = b_frag_t(We_t + (size_t)col * EE + ks * 32, lane);
        }
    // load q tile (vectorized float4)
    {
        int r = t >> 4, c4 = (t & 15) * 4;
        *(float4*)&q_s[r][c4] =
            *(const float4*)&q_f[(size_t)(b * LL + q0 + r) * DD + h * CC + c4];
    }

    const int qq = t >> 4, kk = t & 15;     // this thread's (query,key) pair in tile
    const int cb = kk * 4;                  // 4 output channels owned by this thread
    const int rowbase = t & ~15;
    const int lbase = lane & 16;            // first lane of this query-row's half-wave
    float m_run = -1e30f, l_run = 0.f;
    float out_acc[4] = {0.f, 0.f, 0.f, 0.f};

    for (int kt = 0; kt < 8; ++kt) {
        const int k0 = kt * 16;
        // load kv tile (vectorized)
        {
            int r = t >> 4, c4 = (t & 15) * 4;
            *(float4*)&kv_s[r][c4] =
                *(const float4*)&kv_f[(size_t)(b * LL + k0 + r) * DD + h * CC + c4];
        }
        __syncthreads();   // kv_s ready (q_s on first iter)
        // WMMA accumulator init = kv broadcast: C/D row (roff+r) maps to key roff+r,
        // identical for both M-subtiles, so build once per N-subtile.
        v8f kvinit[4];
#pragma unroll
        for (int ns = 0; ns < 4; ++ns) {
            int c = ns * 16 + (lane & 15);
#pragma unroll
            for (int r = 0; r < 8; ++r) kvinit[ns][r] = kv_s[roff + r][c];
        }
        v8f acc[2][4];
#pragma unroll
        for (int ms = 0; ms < 2; ++ms)
#pragma unroll
            for (int ns = 0; ns < 4; ++ns) acc[ms][ns] = kvinit[ns];
        // kve GEMM: 256 (q,k)-pair rows, A direct from global e_h
#pragma unroll
        for (int ms = 0; ms < 2; ++ms) {
            int m = wave * 32 + ms * 16 + (lane & 15);
            const _Float16* arow =
                e_h + ((size_t)(b * LL + q0 + (m >> 4)) * LL + (k0 + (m & 15))) * EE;
#pragma unroll
            for (int ks = 0; ks < 2; ++ks) {
                v16h af = a_frag_from(arow + ks * 32, lane);
#pragma unroll
                for (int ns = 0; ns < 4; ++ns)
                    acc[ms][ns] = wmma_f16(af, bfr[ns][ks], acc[ms][ns]);
            }
        }
        // spill kve to LDS as f16
#pragma unroll
        for (int ms = 0; ms < 2; ++ms) {
            int rbase = wave * 32 + ms * 16 + roff;
#pragma unroll
            for (int ns = 0; ns < 4; ++ns) {
                int c = ns * 16 + (lane & 15);
#pragma unroll
                for (int r = 0; r < 8; ++r)
                    ee_s[rbase + r][c] = (_Float16)acc[ms][ns][r];
            }
        }
        __syncthreads();   // kve tile ready
        // score for this (q,k) pair (vectorized b128 LDS reads)
        float s = 0.f;
#pragma unroll
        for (int c8 = 0; c8 < 8; ++c8) {
            U8H w; w.u = *(const uint4*)&ee_s[t][c8 * 8];
            const float4 qa = *(const float4*)&q_s[qq][c8 * 8];
            const float4 qb = *(const float4*)&q_s[qq][c8 * 8 + 4];
            s += qa.x * (float)w.h[0] + qa.y * (float)w.h[1] +
                 qa.z * (float)w.h[2] + qa.w * (float)w.h[3] +
                 qb.x * (float)w.h[4] + qb.y * (float)w.h[5] +
                 qb.z * (float)w.h[6] + qb.w * (float)w.h[7];
        }
        s *= 0.125f;                               // 1/sqrt(C)
        s = (s > 0.f) ? s : 0.2f * s;              // leaky relu
        if (adj[(size_t)(b * LL + q0 + qq) * LL + k0 + kk] == 0) s = -1e9f;
        // online softmax over this row's 16 keys (half-wave reductions)
        float mt = s;
#pragma unroll
        for (int o = 1; o < 16; o <<= 1) mt = fmaxf(mt, __shfl_xor(mt, o, 32));
        float m_new = fmaxf(m_run, mt);
        float scale = __expf(m_run - m_new);
        float p = __expf(s - m_new);
        float ps = p;
#pragma unroll
        for (int o = 1; o < 16; o <<= 1) ps += __shfl_xor(ps, o, 32);
        l_run = l_run * scale + ps;
        m_run = m_new;
#pragma unroll
        for (int j = 0; j < 4; ++j) out_acc[j] *= scale;
        // weighted sum: out[q, cb..cb+3] += sum_k p_k * kve[q,k,c]
#pragma unroll
        for (int k2 = 0; k2 < 16; ++k2) {
            float ak = __shfl(p, lbase + k2, 32);
            U4H er; er.u = *(const uint2*)&ee_s[rowbase + k2][cb];
            out_acc[0] += ak * (float)er.h[0];
            out_acc[1] += ak * (float)er.h[1];
            out_acc[2] += ak * (float)er.h[2];
            out_acc[3] += ak * (float)er.h[3];
        }
        __syncthreads();   // protect ee_s/kv_s before next tile
    }
    float inv = 1.f / l_run;
    _Float16* op = attn_h + (size_t)(b * LL + q0 + qq) * DD + h * CC + cb;
#pragma unroll
    for (int j = 0; j < 4; ++j) op[j] = (_Float16)(out_acc[j] * inv);
}

// ---- bias + LayerNorm + ReLU, one row (D=512) per 256-thread block ----
__global__ __launch_bounds__(256) void ln_relu(const float* __restrict__ X,
                                               const float* __restrict__ bf,
                                               const float* __restrict__ gamma,
                                               const float* __restrict__ beta,
                                               float* __restrict__ out) {
    __shared__ float red[16];
    const int row = blockIdx.x, t = threadIdx.x;
    float v0 = X[(size_t)row * DD + t] + bf[t];
    float v1 = X[(size_t)row * DD + t + 256] + bf[t + 256];
    float s = v0 + v1, sq = v0 * v0 + v1 * v1;
#pragma unroll
    for (int o = 1; o < 32; o <<= 1) {
        s  += __shfl_xor(s, o, 32);
        sq += __shfl_xor(sq, o, 32);
    }
    const int wave = t >> 5, lane = t & 31;
    if (lane == 0) { red[wave] = s; red[8 + wave] = sq; }
    __syncthreads();
    float S = 0.f, SQ = 0.f;
#pragma unroll
    for (int i = 0; i < 8; ++i) { S += red[i]; SQ += red[8 + i]; }
    float mu = S * (1.f / 512.f);
    float var = SQ * (1.f / 512.f) - mu * mu;
    float rinv = rsqrtf(var + 1e-5f);
    float o0 = (v0 - mu) * rinv * gamma[t] + beta[t];
    float o1 = (v1 - mu) * rinv * gamma[t + 256] + beta[t + 256];
    out[(size_t)row * DD + t]       = fmaxf(o0, 0.f);
    out[(size_t)row * DD + t + 256] = fmaxf(o1, 0.f);
}

extern "C" void kernel_launch(void* const* d_in, const int* in_sizes, int n_in,
                              void* d_out, int out_size, void* d_ws, size_t ws_size,
                              hipStream_t stream) {
    const float* x     = (const float*)d_in[0];
    const int*   adj   = (const int*)  d_in[1];
    const float* e     = (const float*)d_in[2];
    const float* Wq    = (const float*)d_in[3];
    const float* Wkv   = (const float*)d_in[4];
    const float* We    = (const float*)d_in[5];
    const float* Wf    = (const float*)d_in[6];
    const float* bf    = (const float*)d_in[7];
    const float* gamma = (const float*)d_in[8];
    const float* beta  = (const float*)d_in[9];
    float* out = (float*)d_out;

    const int BL = BB * LL;               // 1024
    const int nX  = BL * DD;              // 524288
    const int nW  = DD * DD;              // 262144
    const int nWe = EE * DD;              // 32768
    const int nE  = BB * LL * LL * EE;    // 8388608

    char* ws = (char*)d_ws;
    size_t off = 0;
    auto alloc = [&](size_t bytes) -> void* {
        void* p = ws + off;
        off += (bytes + 255) & ~(size_t)255;
        return p;
    };
    _Float16* x_h    = (_Float16*)alloc((size_t)nX * 2);
    _Float16* Wq_t   = (_Float16*)alloc((size_t)nW * 2);   // transposed (N x K)
    _Float16* Wkv_t  = (_Float16*)alloc((size_t)nW * 2);
    _Float16* Wf_t   = (_Float16*)alloc((size_t)nW * 2);
    _Float16* We_t   = (_Float16*)alloc((size_t)nWe * 2);  // (D x E)
    _Float16* e_h    = (_Float16*)alloc((size_t)nE * 2);
    float*    q_f    = (float*)   alloc((size_t)nX * 4);
    float*    kv_f   = (float*)   alloc((size_t)nX * 4);
    _Float16* attn_h = (_Float16*)alloc((size_t)nX * 2);
    float*    ffn_f  = (float*)   alloc((size_t)nX * 4);

    // 1) conversions (weights transposed for contiguous B fragments)
    cvt_f32_f16<<<(nX + 255) / 256, 256, 0, stream>>>(x, x_h, nX);
    cvt_f32_f16<<<(nE + 255) / 256, 256, 0, stream>>>(e, e_h, nE);
    cvt_transpose_f16<<<(nW + 255) / 256, 256, 0, stream>>>(Wq,  Wq_t,  DD, DD);
    cvt_transpose_f16<<<(nW + 255) / 256, 256, 0, stream>>>(Wkv, Wkv_t, DD, DD);
    cvt_transpose_f16<<<(nW + 255) / 256, 256, 0, stream>>>(Wf,  Wf_t,  DD, DD);
    cvt_transpose_f16<<<(nWe + 255) / 256, 256, 0, stream>>>(We, We_t,  EE, DD);

    // 2) q = x @ Wq, kv = x @ Wkv   (M=1024, N=512, K=512)
    dim3 ggrid(DD / 64, BL / 128);
    gemm_f16<<<ggrid, 256, 0, stream>>>(x_h, Wq_t,  q_f,  BL, DD, DD);
    gemm_f16<<<ggrid, 256, 0, stream>>>(x_h, Wkv_t, kv_f, BL, DD, DD);

    // 3) fused edge-modulated attention
    dim3 agrid(LL / 16, HH, BB);
    attn_fused<<<agrid, 256, 0, stream>>>(e_h, We_t, q_f, kv_f, adj, attn_h);

    // 4) FFN GEMM: attn @ Wf
    gemm_f16<<<ggrid, 256, 0, stream>>>(attn_h, Wf_t, ffn_f, BL, DD, DD);

    // 5) bias + LayerNorm + ReLU
    ln_relu<<<BL, 256, 0, stream>>>(ffn_f, bf, gamma, beta, out);
}
